// CQT_85856396247287
// MI455X (gfx1250) — matrix operand order repaired and checked by reading, not compiled, not run on hardware
//
#include <hip/hip_runtime.h>
#include <hip/hip_bf16.h>

// ---------------- CQT configuration (matches reference) ----------------
#define SRATE     44100
#define HOP       484
#define SIG_LEN   220500
#define N_FRAMES  456
#define N_BINS    252
#define MAX_LEN   69365          // ceil(Q*SR/FMIN), Q = 1/(2^(1/36)-1)
#define CENTER    34682          // MAX_LEN/2
#define FMIN_F    32.70319566257483f
#define TWO_PI_F  6.28318530717958647692f

// ---------------- tiling ----------------
#define KPAD      69632          // 2176*32 >= MAX_LEN; tail windows to zero
#define NZ        8              // K-split chunks
#define KCHUNK    (KPAD / NZ)    // 8704 = 272 * 32
#define MT64      8              // ceil(456/64) frame super-tiles (64 frames/wave)
#define M_PAD     (MT64 * 64)    // 512
#define NBLK      2              // 2 * (8 waves * 16) = 256 padded bins

// ---------------- workspace layout (bytes) ----------------
#define PAD_ELEMS 316960         // >= 511*484 + KPAD = 316956, multiple of 4
#define OFF_PADSIG 0
#define OFF_NORM   (PAD_ELEMS * 2)          // 633920 (4-aligned)
#define OFF_PART   (OFF_NORM + 256 * 4)     // partials: [NZ][M_PAD][256][2] f32

typedef __attribute__((ext_vector_type(16))) __bf16             v16bf;
typedef __attribute__((ext_vector_type(8)))  unsigned int       v8u;
typedef __attribute__((ext_vector_type(4)))  unsigned long long v4u64;
typedef __attribute__((ext_vector_type(8)))  float              v8f;

__device__ __forceinline__ unsigned int bfbits(float x) {       // rounded, in [31:16]
    unsigned int u = __float_as_uint(x);
    return u + 0x7fffu + ((u >> 16) & 1u);
}
__device__ __forceinline__ unsigned int packbf(float lo, float hi) {
    return (bfbits(lo) >> 16) | (bfbits(hi) & 0xffff0000u);
}
__device__ __forceinline__ unsigned short f2bf(float x) {
    return (unsigned short)(bfbits(x) >> 16);
}

// ---- phase 0: signal -> zero-padded bf16 buffer (center framing baked in) ----
__global__ void cqt_pad(const float* __restrict__ sig, void* wsv) {
    unsigned short* p = (unsigned short*)((char*)wsv + OFF_PADSIG);
    int i = blockIdx.x * 256 + threadIdx.x;
    if (i >= PAD_ELEMS) return;
    int s = i - CENTER;
    float v = (s >= 0 && s < SIG_LEN) ? sig[s] : 0.0f;
    p[i] = f2bf(v);
}

// ---- phase 1: per-bin inverse norm 1 / (sum(hann) * sqrt(L)) ----
__global__ __launch_bounds__(256) void cqt_norm(void* wsv) {
    float* invn = (float*)((char*)wsv + OFF_NORM);
    const int bin = blockIdx.x;
    __shared__ float red[256];
    if (bin >= N_BINS) { if (threadIdx.x == 0) invn[bin] = 0.0f; return; }
    const double Qd = 1.0 / (exp2(1.0 / 36.0) - 1.0);
    float f     = FMIN_F * exp2f((float)bin * (1.0f / 36.0f));
    float L     = (float)(Qd * (double)SRATE) / f;
    float halfL = 0.5f * L;
    float winc  = TWO_PI_F / L;
    float s = 0.0f;
    for (int t = threadIdx.x; t < MAX_LEN; t += 256) {
        float tt = (float)(t - CENTER);
        float cw = __cosf(winc * tt);                  // unconditional; select after
        float m  = (fabsf(tt) < halfL) ? 1.0f : 0.0f;
        s += (0.5f + 0.5f * cw) * m;
    }
    red[threadIdx.x] = s;
    __syncthreads();
    for (int st = 128; st > 0; st >>= 1) {
        if (threadIdx.x < st) red[threadIdx.x] += red[threadIdx.x + st];
        __syncthreads();
    }
    if (threadIdx.x == 0) invn[bin] = 1.0f / (red[0] * sqrtf(L));
}

// ---- phase 2: fused bank-generation + bf16 WMMA GEMM (real & imag) ----
// Each wave: 16 bins x 64 frames (4 M-subtiles share one generated B pair).
__global__ __launch_bounds__(256) void cqt_gemm(void* __restrict__ wsv) {
    char* wsb = (char*)wsv;
    const unsigned long long* sig64 = (const unsigned long long*)(wsb + OFF_PADSIG);
    const float* invn = (const float*)(wsb + OFF_NORM);
    float* part = (float*)(wsb + OFF_PART);

    const int lane  = threadIdx.x & 31;
    const int wave  = threadIdx.x >> 5;
    const int mbase = blockIdx.x * 64;                 // 64-frame super-tile
    const int nbase = blockIdx.y * 128 + wave * 16;    // bin tile (one per wave)
    const int z     = blockIdx.z;
    const int k0    = z * KCHUNK;

    // per-lane invariants
    const int l15  = lane & 15;
    const int aks  = (lane < 16) ? 0 : 8;              // A-fragment K sub-offset
    const int bks  = (lane < 16) ? 0 : 8;              // B-fragment K sub-offset
    const int bin  = nbase + l15;
    int mrow[4];
#pragma unroll
    for (int i = 0; i < 4; ++i) mrow[i] = (mbase + i * 16 + l15) * HOP;

    const double Qd = 1.0 / (exp2(1.0 / 36.0) - 1.0);
    const float QSR   = (float)(Qd * (double)SRATE);
    const float f     = FMIN_F * exp2f((float)bin * (1.0f / 36.0f));
    const float L     = QSR / f;
    const float halfL = 0.5f * L;
    const float w2    = TWO_PI_F * f * (1.0f / (float)SRATE);
    const float winc  = TWO_PI_F / L;
    const float hinv  = 0.5f * invn[bin];              // folds the 0.5 of Hann

    // wave-uniform K clipping to the widest Hann support in this bin tile;
    // readfirstlane tells the compiler these are scalar -> SALU loop, no EXEC games
    const float Lmax = QSR / (FMIN_F * exp2f((float)nbase * (1.0f / 36.0f)));
    int kmin = CENTER - (int)(0.5f * Lmax) - 2;
    if (kmin < 0) kmin = 0;
    int kmax = CENTER + (int)(0.5f * Lmax) + 2;
    int ks = kmin & ~31;         if (ks < k0) ks = k0;
    int ke = (kmax + 32) & ~31;  if (ke > k0 + KCHUNK) ke = k0 + KCHUNK;
    ks = __builtin_amdgcn_readfirstlane(ks);
    ke = __builtin_amdgcn_readfirstlane(ke);

    v8f accr[4], acci[4];
#pragma unroll
    for (int i = 0; i < 4; ++i) {
        accr[i] = (v8f){0.f, 0.f, 0.f, 0.f, 0.f, 0.f, 0.f, 0.f};
        acci[i] = (v8f){0.f, 0.f, 0.f, 0.f, 0.f, 0.f, 0.f, 0.f};
    }

    for (int kb = ks; kb < ke; kb += 32) {
        // ---- B fragments: K weights bks+{0..7}, bks+{16..23} of this lane's bin column ----
        const float tf = (float)(kb + bks - CENTER);   // one cvt per iteration
        v8u brp, bip;
#pragma unroll
        for (int p = 0; p < 8; ++p) {
            float t0 = tf + (float)(2 * p + (p >= 4 ? 8 : 0));
            float t1 = tf + (float)(2 * p + 1 + (p >= 4 ? 8 : 0));
            // unconditional cos; branch-free mask select (v_cndmask, EXEC untouched)
            float cw0 = __cosf(winc * t0);
            float cw1 = __cosf(winc * t1);
            float m0  = (fabsf(t0) < halfL) ? hinv : 0.0f;
            float m1  = (fabsf(t1) < halfL) ? hinv : 0.0f;
            float w0  = (1.0f + cw0) * m0;
            float w1  = (1.0f + cw1) * m1;
            float s0, c0, s1, c1;
            __sincosf(w2 * t0, &s0, &c0);
            __sincosf(w2 * t1, &s1, &c1);
            brp[p] = packbf(w0 * c0, w1 * c1);
            bip[p] = packbf(w0 * s0, w1 * s1);
        }
        v16bf vbr = __builtin_bit_cast(v16bf, brp);
        v16bf vbi = __builtin_bit_cast(v16bf, bip);

        // ---- 4 A fragments (already in WMMA bf16 register layout) + 8 WMMAs ----
#pragma unroll
        for (int i = 0; i < 4; ++i) {
            int abase = mrow[i] + kb + aks;            // multiple of 4 -> 8B aligned
            v4u64 aq;
            aq.x = sig64[(abase >> 2) + 0];            // K + {0..3}
            aq.y = sig64[(abase >> 2) + 1];            // K + {4..7}
            aq.z = sig64[(abase >> 2) + 4];            // K + {16..19}
            aq.w = sig64[(abase >> 2) + 5];            // K + {20..23}
            v16bf a = __builtin_bit_cast(v16bf, aq);
            accr[i] = __builtin_amdgcn_wmma_f32_16x16x32_bf16(
                          false, a, false, vbr, (short)0, accr[i], false, false);
            acci[i] = __builtin_amdgcn_wmma_f32_16x16x32_bf16(
                          false, a, false, vbi, (short)0, acci[i], false, false);
        }
    }

    // C/D layout: lanes 0-15 VGPR r -> M=r; lanes 16-31 VGPR r -> M=8+r; N = lane&15
    const int mloc = (lane < 16) ? 0 : 8;
#pragma unroll
    for (int i = 0; i < 4; ++i) {
#pragma unroll
        for (int r = 0; r < 8; ++r) {
            int mm = mbase + i * 16 + mloc + r;
            size_t o = ((((size_t)z * M_PAD) + mm) * 256 + bin) * 2;
            part[o]     = accr[i][r];
            part[o + 1] = acci[i][r];
        }
    }
}

// ---- phase 3: deterministic K-chunk reduce + magnitude ----
__global__ void cqt_finalize(const void* wsv, float* __restrict__ out) {
    const float* part = (const float*)((const char*)wsv + OFF_PART);
    int i = blockIdx.x * 256 + threadIdx.x;
    if (i >= N_BINS * N_FRAMES) return;
    int b  = i / N_FRAMES;
    int fr = i - b * N_FRAMES;
    float cr = 0.0f, ci = 0.0f;
#pragma unroll
    for (int zz = 0; zz < NZ; ++zz) {
        size_t o = ((((size_t)zz * M_PAD) + fr) * 256 + b) * 2;
        cr += part[o];
        ci += part[o + 1];
    }
    out[i] = sqrtf(cr * cr + ci * ci);   // out[b*456 + fr], matches [252,456]
}

extern "C" void kernel_launch(void* const* d_in, const int* in_sizes, int n_in,
                              void* d_out, int out_size, void* d_ws, size_t ws_size,
                              hipStream_t stream) {
    (void)in_sizes; (void)n_in; (void)out_size; (void)ws_size;
    const float* signal = (const float*)d_in[0];
    float* out = (float*)d_out;

    cqt_pad<<<(PAD_ELEMS + 255) / 256, 256, 0, stream>>>(signal, d_ws);
    cqt_norm<<<256, 256, 0, stream>>>(d_ws);
    cqt_gemm<<<dim3(MT64, NBLK, NZ), 256, 0, stream>>>(d_ws);
    cqt_finalize<<<(N_BINS * N_FRAMES + 255) / 256, 256, 0, stream>>>(d_ws, out);
}
